// ModalGAT_28089086116174
// MI455X (gfx1250) — matrix operand order repaired and checked by reading, not compiled, not run on hardware
//
#include <hip/hip_runtime.h>
#include <math.h>

typedef __attribute__((ext_vector_type(16))) _Float16 v16h;
typedef __attribute__((ext_vector_type(8)))  float    v8f;
typedef __attribute__((ext_vector_type(4)))  int      v4i;

#define IN_DIM   128
#define HC       256     // HEADS * OUT_DIM
#define HEADS    4
#define CDIM     64
#define NEG_SLOPE 0.2f

#if __has_builtin(__builtin_amdgcn_global_load_async_to_lds_b128)
#define HAVE_ASYNC_LDS 1
#else
#define HAVE_ASYNC_LDS 0
#endif

// ---------------------------------------------------------------------------
// float atomic max via int/uint monotonic trick (amax initialized to -inf)
// ---------------------------------------------------------------------------
__device__ __forceinline__ void atomicMaxF32(float* addr, float val) {
    if (val >= 0.0f) {
        atomicMax(reinterpret_cast<int*>(addr), __float_as_int(val));
    } else {
        atomicMin(reinterpret_cast<unsigned int*>(addr), __float_as_uint(val));
    }
}

// ---------------------------------------------------------------------------
// 0) init accumulators: amax=-inf, denom=0, outacc=0
// ---------------------------------------------------------------------------
__global__ __launch_bounds__(256)
void gat_init_kernel(float* __restrict__ amax, float* __restrict__ denom,
                     float* __restrict__ outacc, int N)
{
    int idx = blockIdx.x * blockDim.x + threadIdx.x;
    if (idx < N * HC) outacc[idx] = 0.0f;
    if (idx < N * HEADS) {
        amax[idx]  = -INFINITY;
        denom[idx] = 0.0f;
    }
}

// ---------------------------------------------------------------------------
// 0b) prepack W (f32, [128,256] row-major) into f16 B-fragments in exact
//     WMMA register order. frag f = ntile*4 + kc; within a frag, lane l
//     (grp = l>>4, li = l&15) holds 16 halfs: slot s -> W[kc*32+grp*16+s][ntile*16+li].
//     64 frags * 32 lanes * 16 halfs = 64 KB, L2-resident, reused by all blocks.
// ---------------------------------------------------------------------------
__global__ __launch_bounds__(256)
void gat_wpack_kernel(const float* __restrict__ W, _Float16* __restrict__ wpack)
{
    const int t = blockIdx.x * blockDim.x + threadIdx.x;
    if (t >= 16 * 4 * 32) return;
    const int lane  = t & 31;
    const int f     = t >> 5;        // 0..63
    const int kc    = f & 3;
    const int ntile = f >> 2;
    const int grp   = lane >> 4;
    const int li    = lane & 15;
    const int n     = ntile * 16 + li;
    const int kbase = kc * 32 + grp * 16;
    v16h v;
    #pragma unroll
    for (int s = 0; s < 16; ++s)
        v[s] = (_Float16)W[(size_t)(kbase + s) * HC + n];
    *reinterpret_cast<v16h*>(wpack + (size_t)(f * 32 + lane) * 16) = v;
}

// ---------------------------------------------------------------------------
// 1) h = x @ W  via v_wmma_f32_16x16x32_f16
//    grid = (N/16, 2), block = 256 (8 waves). Block stages a 16x128 f32 A-tile
//    into LDS (row pad 132 to avoid stride-128 bank conflicts) — via
//    global_load_async_to_lds_b128 when available. Wave w computes the 16x16
//    output tile at columns blockIdx.y*128 + w*16, chaining 4 WMMAs over K=128;
//    B fragments come prepacked (contiguous b128 loads).
// ---------------------------------------------------------------------------
__global__ __launch_bounds__(256)
void gat_gemm_kernel(const float* __restrict__ x, const _Float16* __restrict__ wpack,
                     float* __restrict__ h, int N)
{
    __shared__ float lds[16][132];
    const int tid   = threadIdx.x;
    const int mbase = blockIdx.x * 16;

    // stage A tile (16 rows x 128 cols, f32) into LDS
    {
        const int r = tid >> 4;           // 0..15
        const int c = (tid & 15) * 8;     // 0..120
        int row = mbase + r;
        if (row >= N) row = N - 1;        // clamp (harmless dup, stores guarded)
        const float* src = x + (size_t)row * IN_DIM + c;
#if HAVE_ASYNC_LDS
        __builtin_amdgcn_global_load_async_to_lds_b128(
            (__attribute__((address_space(1))) v4i*)(src),
            (__attribute__((address_space(3))) v4i*)&lds[r][c], 0, 0);
        __builtin_amdgcn_global_load_async_to_lds_b128(
            (__attribute__((address_space(1))) v4i*)(src + 4),
            (__attribute__((address_space(3))) v4i*)&lds[r][c + 4], 0, 0);
#if __has_builtin(__builtin_amdgcn_s_wait_asynccnt)
        __builtin_amdgcn_s_wait_asynccnt(0);
#else
        asm volatile("s_wait_asynccnt 0x0" ::: "memory");
#endif
#else
        const float4* s4 = reinterpret_cast<const float4*>(src);
        float4 p0 = s4[0];
        float4 p1 = s4[1];
        lds[r][c+0] = p0.x; lds[r][c+1] = p0.y; lds[r][c+2] = p0.z; lds[r][c+3] = p0.w;
        lds[r][c+4] = p1.x; lds[r][c+5] = p1.y; lds[r][c+6] = p1.z; lds[r][c+7] = p1.w;
#endif
    }
    __syncthreads();

    const int wave  = tid >> 5;
    const int lane  = tid & 31;
    const int grp   = lane >> 4;      // lane group 0/1
    const int li    = lane & 15;
    const int ntile = blockIdx.y * 8 + wave;
    const int nbase = ntile * 16;

    v8f acc = {};
    #pragma unroll
    for (int kc = 0; kc < 4; ++kc) {
        const int kbase = kc * 32;

        // A fragment (16-bit A 16x32): slots 0..15 -> k = 16*grp + s
        //                              (matches prepacked B order)
        v16h a;
        {
            const float* ar = &lds[li][kbase + grp * 16];
            #pragma unroll
            for (int s = 0; s < 8; ++s) a[s]     = (_Float16)ar[s];
            #pragma unroll
            for (int s = 0; s < 8; ++s) a[8 + s] = (_Float16)ar[8 + s];
        }

        // B fragment: prepacked in register order -> contiguous 32B load
        v16h b = *reinterpret_cast<const v16h*>(
            wpack + (size_t)((ntile * 4 + kc) * 32 + lane) * 16);

        acc = __builtin_amdgcn_wmma_f32_16x16x32_f16(
            /*neg_a=*/false, a, /*neg_b=*/false, b,
            /*c_mod=*/(short)0, acc, /*reuse_a=*/false, /*reuse_b=*/false);
    }

    // D layout: VGPR r -> row r + 8*grp, col = li
    #pragma unroll
    for (int r = 0; r < 8; ++r) {
        const int row = mbase + grp * 8 + r;
        if (row < N)
            h[(size_t)row * HC + nbase + li] = acc[r];
    }
}

// ---------------------------------------------------------------------------
// 2) per-node attention projections a_src[n,h], a_dst[n,h]
// ---------------------------------------------------------------------------
__global__ __launch_bounds__(256)
void gat_att_kernel(const float* __restrict__ h, const float* __restrict__ att_src,
                    const float* __restrict__ att_dst, float* __restrict__ a_src,
                    float* __restrict__ a_dst, int N)
{
    int idx = blockIdx.x * blockDim.x + threadIdx.x;    // n*HEADS + head
    if (idx >= N * HEADS) return;
    const int head = idx & (HEADS - 1);
    const int n    = idx >> 2;
    const float* hr = h + (size_t)n * HC + head * CDIM;
    const float* as = att_src + head * CDIM;
    const float* ad = att_dst + head * CDIM;
    float ss = 0.0f, sd = 0.0f;
    #pragma unroll 8
    for (int c = 0; c < CDIM; ++c) {
        float v = hr[c];
        ss += v * as[c];
        sd += v * ad[c];
    }
    a_src[idx] = ss;
    a_dst[idx] = sd;
}

// ---------------------------------------------------------------------------
// 3) edge logits + leaky-relu + segment max (edges E..E+N-1 are self loops)
// ---------------------------------------------------------------------------
__global__ __launch_bounds__(256)
void gat_edge_max_kernel(const int* __restrict__ ei, int E, int N,
                         const float* __restrict__ a_src, const float* __restrict__ a_dst,
                         float* __restrict__ alpha, float* __restrict__ amax)
{
    int e = blockIdx.x * blockDim.x + threadIdx.x;
    const int Et = E + N;
    if (e >= Et) return;
    const int s = (e < E) ? ei[e]     : (e - E);
    const int d = (e < E) ? ei[E + e] : (e - E);
    #pragma unroll
    for (int hd = 0; hd < HEADS; ++hd) {
        float al = a_src[s * HEADS + hd] + a_dst[d * HEADS + hd];
        al = (al >= 0.0f) ? al : NEG_SLOPE * al;
        alpha[(size_t)e * HEADS + hd] = al;
        atomicMaxF32(&amax[d * HEADS + hd], al);
    }
}

// ---------------------------------------------------------------------------
// 4) e = exp(alpha - amax[dst]); segment-sum into denom; store e in alpha
// ---------------------------------------------------------------------------
__global__ __launch_bounds__(256)
void gat_edge_exp_kernel(const int* __restrict__ ei, int E, int N,
                         float* __restrict__ alpha, const float* __restrict__ amax,
                         float* __restrict__ denom)
{
    int e = blockIdx.x * blockDim.x + threadIdx.x;
    const int Et = E + N;
    if (e >= Et) return;
    const int d = (e < E) ? ei[E + e] : (e - E);
    #pragma unroll
    for (int hd = 0; hd < HEADS; ++hd) {
        float ev = __expf(alpha[(size_t)e * HEADS + hd] - amax[d * HEADS + hd]);
        alpha[(size_t)e * HEADS + hd] = ev;
        atomicAdd(&denom[d * HEADS + hd], ev);
    }
}

// ---------------------------------------------------------------------------
// 5) weighted scatter: out[dst,h,c] += (e/denom) * hmsg[src,h,c]
//    one wave32 per edge; lane owns 8 contiguous channels (head = lane>>3);
//    coalesced 1KB b128 gather of h[src], f32 atomics into outacc[dst].
// ---------------------------------------------------------------------------
__global__ __launch_bounds__(256)
void gat_scatter_kernel(const int* __restrict__ ei, int E, int N,
                        const float* __restrict__ h, const float* __restrict__ alpha,
                        const float* __restrict__ denom, float* __restrict__ outacc)
{
    const int wave = (blockIdx.x * blockDim.x + threadIdx.x) >> 5;
    const int lane = threadIdx.x & 31;
    const int Et = E + N;
    if (wave >= Et) return;
    const int e = wave;
    const int s = (e < E) ? ei[e]     : (e - E);
    const int d = (e < E) ? ei[E + e] : (e - E);

    const int head = lane >> 3;                         // 8 lanes per head
    const float w = alpha[(size_t)e * HEADS + head] / denom[d * HEADS + head];

    const float4* hs = reinterpret_cast<const float4*>(h + (size_t)s * HC + lane * 8);
    float4 p0 = hs[0];
    float4 p1 = hs[1];
    float* ob = outacc + (size_t)d * HC + lane * 8;
    atomicAdd(ob + 0, w * p0.x);
    atomicAdd(ob + 1, w * p0.y);
    atomicAdd(ob + 2, w * p0.z);
    atomicAdd(ob + 3, w * p0.w);
    atomicAdd(ob + 4, w * p1.x);
    atomicAdd(ob + 5, w * p1.y);
    atomicAdd(ob + 6, w * p1.z);
    atomicAdd(ob + 7, w * p1.w);
}

// ---------------------------------------------------------------------------
// 6) mean over heads + bias + LayerNorm + PReLU. One wave32 per node,
//    each lane owns channels (lane, lane+32); wave32 shfl_xor reductions.
// ---------------------------------------------------------------------------
__global__ __launch_bounds__(256)
void gat_final_kernel(const float* __restrict__ outacc, const float* __restrict__ bias,
                      const float* __restrict__ gamma, const float* __restrict__ beta,
                      const float* __restrict__ prelu_w, float* __restrict__ out, int N)
{
    const int wave = (blockIdx.x * blockDim.x + threadIdx.x) >> 5;
    const int lane = threadIdx.x & 31;
    if (wave >= N) return;
    const int n = wave;
    const float* ob = outacc + (size_t)n * HC;
    const int c0 = lane, c1 = lane + 32;

    float v0 = 0.0f, v1 = 0.0f;
    #pragma unroll
    for (int hd = 0; hd < HEADS; ++hd) {
        v0 += ob[hd * CDIM + c0];
        v1 += ob[hd * CDIM + c1];
    }
    v0 = v0 * 0.25f + bias[c0];
    v1 = v1 * 0.25f + bias[c1];

    float ssum = v0 + v1;
    float ssq  = v0 * v0 + v1 * v1;
    #pragma unroll
    for (int off = 16; off >= 1; off >>= 1) {
        ssum += __shfl_xor(ssum, off, 32);
        ssq  += __shfl_xor(ssq,  off, 32);
    }
    const float mean = ssum * (1.0f / 64.0f);
    const float var  = ssq * (1.0f / 64.0f) - mean * mean;
    const float inv  = rsqrtf(var + 1e-5f);
    const float pw   = prelu_w[0];

    float o0 = (v0 - mean) * inv * gamma[c0] + beta[c0];
    float o1 = (v1 - mean) * inv * gamma[c1] + beta[c1];
    out[(size_t)n * CDIM + c0] = (o0 >= 0.0f) ? o0 : pw * o0;
    out[(size_t)n * CDIM + c1] = (o1 >= 0.0f) ? o1 : pw * o1;
}

// ---------------------------------------------------------------------------
// launch
// ---------------------------------------------------------------------------
extern "C" void kernel_launch(void* const* d_in, const int* in_sizes, int n_in,
                              void* d_out, int out_size, void* d_ws, size_t ws_size,
                              hipStream_t stream)
{
    const float* x        = (const float*)d_in[0];
    const int*   ei       = (const int*)  d_in[1];
    const float* W        = (const float*)d_in[2];
    const float* att_src  = (const float*)d_in[3];
    const float* att_dst  = (const float*)d_in[4];
    const float* bias     = (const float*)d_in[5];
    const float* gamma    = (const float*)d_in[6];
    const float* beta     = (const float*)d_in[7];
    const float* prelu_w  = (const float*)d_in[8];
    float*       out      = (float*)d_out;

    const int N  = in_sizes[0] / IN_DIM;
    const int E  = in_sizes[1] / 2;
    const int Et = E + N;

    // workspace layout (floats); wpack first to keep 32B alignment
    float* ws       = (float*)d_ws;
    _Float16* wpack = (_Float16*)ws;                     // 32768 halfs = 16384 floats
    float* hbuf     = ws + 16384;                        // N*256
    float* a_src    = hbuf   + (size_t)N * HC;           // N*4
    float* a_dst    = a_src  + (size_t)N * HEADS;        // N*4
    float* amax     = a_dst  + (size_t)N * HEADS;        // N*4
    float* denom    = amax   + (size_t)N * HEADS;        // N*4
    float* outacc   = denom  + (size_t)N * HEADS;        // N*256
    float* alpha    = outacc + (size_t)N * HC;           // Et*4

    // 0) init accumulators + prepack W fragments
    gat_init_kernel<<<(N * HC + 255) / 256, 256, 0, stream>>>(amax, denom, outacc, N);
    gat_wpack_kernel<<<8, 256, 0, stream>>>(W, wpack);

    // 1) WMMA GEMM: h = f16(x) @ f16(W), f32 accumulate
    {
        dim3 grid((N + 15) / 16, 2);
        gat_gemm_kernel<<<grid, 256, 0, stream>>>(x, wpack, hbuf, N);
    }

    // 2) attention projections
    gat_att_kernel<<<(N * HEADS + 255) / 256, 256, 0, stream>>>(
        hbuf, att_src, att_dst, a_src, a_dst, N);

    // 3) edge logits + segment max
    gat_edge_max_kernel<<<(Et + 255) / 256, 256, 0, stream>>>(
        ei, E, N, a_src, a_dst, alpha, amax);

    // 4) exp + segment sum
    gat_edge_exp_kernel<<<(Et + 255) / 256, 256, 0, stream>>>(
        ei, E, N, alpha, amax, denom);

    // 5) weighted scatter-aggregate (wave per edge)
    {
        long long threads = (long long)Et * 32;
        gat_scatter_kernel<<<(unsigned)((threads + 255) / 256), 256, 0, stream>>>(
            ei, E, N, hbuf, alpha, denom, outacc);
    }

    // 6) head-mean + LayerNorm + PReLU (wave per node)
    {
        long long threads = (long long)N * 32;
        gat_final_kernel<<<(unsigned)((threads + 255) / 256), 256, 0, stream>>>(
            outacc, bias, gamma, beta, prelu_w, out, N);
    }
}
